// GlobalAttentionPooling_46720654246226
// MI455X (gfx1250) — matrix operand order repaired and checked by reading, not compiled, not run on hardware
//
#include <hip/hip_runtime.h>
#include <hip/hip_bf16.h>
#include <math.h>

// ---------------- problem constants ----------------
#define N_NODES  500000
#define DIM      256          // NODE_DIM (= K of the GEMM)
#define OUTD     256          // OUT_DIM
#define NGRAPH   4096

// WMMA vector types (probe-confirmed signatures)
typedef __attribute__((ext_vector_type(16))) __bf16 v16bf;
typedef __attribute__((ext_vector_type(8)))  __bf16 v8bf;
typedef __attribute__((ext_vector_type(4)))  __bf16 v4bf;
typedef __attribute__((ext_vector_type(8)))  float  v8f;

__device__ __forceinline__ float fast_rcp(float a) {
    return __builtin_amdgcn_rcpf(a);           // v_rcp_f32 (1 ulp)
}

// ---------------- kernel 1: init ----------------
__global__ void init_kernel(float* __restrict__ z, float* __restrict__ gmax,
                            float* __restrict__ gsum) {
    int i = blockIdx.x * blockDim.x + threadIdx.x;
    if (i < NGRAPH * OUTD) z[i] = 0.0f;
    if (i < NGRAPH) { gmax[i] = -INFINITY; gsum[i] = 0.0f; }
}

// ---------------- kernel 2: pack value_w into WMMA-B bf16 layout ----------------
// B (K x N = 32 x 16 per tile), bf16, wave32:
//   lane half 0 holds K = 0..15, half 1 holds K = 16..31  (element e -> k = e + 16*half)
//   lane n16 = lane & 15 is the column within the 16-wide N tile.
// packedB[((kt*16 + nt)*32 + lane)*16 + e]
__global__ void pack_b_kernel(const float* __restrict__ w, __bf16* __restrict__ pB) {
    int t    = blockIdx.x * blockDim.x + threadIdx.x;   // 0 .. 65535
    int e    = t & 15;
    int lane = (t >> 4) & 31;
    int nt   = (t >> 9) & 15;
    int kt   = t >> 13;
    int half = lane >> 4, n16 = lane & 15;
    int k = kt * 32 + e + 16 * half;
    int n = nt * 16 + n16;
    pB[t] = (__bf16)w[k * OUTD + n];
}

// ---------------- kernel 3: gates = x @ gate_w + b, segment max ----------------
__global__ void gates_max_kernel(const float* __restrict__ x,
                                 const int*   __restrict__ batch,
                                 const float* __restrict__ gate_w,
                                 const float* __restrict__ gate_b,
                                 float* __restrict__ gates,
                                 float* __restrict__ gmax) {
    int lane   = threadIdx.x & 31;
    int wave   = blockIdx.x * (blockDim.x >> 5) + (threadIdx.x >> 5);
    int nwaves = gridDim.x * (blockDim.x >> 5);
    // each lane owns 8 contiguous weights; preload once
    const float4* gw4 = (const float4*)(gate_w + lane * 8);
    float4 w0 = gw4[0], w1 = gw4[1];
    float  gb = gate_b[0];
    for (int row = wave; row < N_NODES; row += nwaves) {
        const float4* xr = (const float4*)(x + (size_t)row * DIM + lane * 8);
        float4 a0 = xr[0], a1 = xr[1];
        float p = a0.x*w0.x + a0.y*w0.y + a0.z*w0.z + a0.w*w0.w
                + a1.x*w1.x + a1.y*w1.y + a1.z*w1.z + a1.w*w1.w;
        #pragma unroll
        for (int off = 16; off >= 1; off >>= 1) p += __shfl_xor(p, off, 32);
        if (lane == 0) {
            float g = p + gb;
            gates[row] = g;
            atomicMax(&gmax[batch[row]], g);   // global_atomic_max_num_f32
        }
    }
}

// ---------------- kernel 4: exp(g - max) + segment sum ----------------
__global__ void exp_sum_kernel(const int* __restrict__ batch,
                               const float* __restrict__ gates,
                               const float* __restrict__ gmax,
                               float* __restrict__ expg,
                               float* __restrict__ gsum) {
    int i = blockIdx.x * blockDim.x + threadIdx.x;
    if (i >= N_NODES) return;
    int g = batch[i];
    float e = __expf(gates[i] - gmax[g]);
    expg[i] = e;
    atomicAdd(&gsum[g], e);                    // global_atomic_add_f32
}

// ---------------- kernel 5: fused value GEMM (bf16 WMMA) + silu + attn + seg-sum ----------------
#define ROWS        32
#define LDS_STRIDE  264   // 256 + 8 bf16 pad: 528 B/row keeps 16-B chunk alignment,
                          // shifts rows by 4 dwords -> conflict-free ds_load_b128

__global__ void __launch_bounds__(256)
fused_pool_kernel(const float* __restrict__ x,
                  const int*   __restrict__ batch,
                  const float* __restrict__ value_b,
                  const __bf16* __restrict__ pB,
                  const float* __restrict__ expg,
                  const float* __restrict__ gsum,
                  float* __restrict__ z,
                  float* __restrict__ attn_out) {
    __shared__ __bf16 xt[ROWS * LDS_STRIDE];
    __shared__ float  s_attn[ROWS];
    __shared__ int    s_g[ROWS];

    int tid  = threadIdx.x;
    int row0 = blockIdx.x * ROWS;

    // --- stage attn + graph ids for this row tile ---
    if (tid < ROWS) {
        int r = row0 + tid;
        int g = batch[r];
        float a = expg[r] * fast_rcp(gsum[g] + 1e-8f);   // gsum >= 1 always
        attn_out[r] = a;
        s_attn[tid] = a;
        s_g[tid]    = g;
    }

    // --- stage x tile: fp32 -> bf16, stored in natural k order ---
    // A layout (16-bit, 16x32): half 0 reads K 0-15, half 1 reads K 16-31 (matches packed B).
    // Store each 32-wide kt block in natural order so each lane's 16
    // elements are one contiguous 32-byte chunk. A 4-aligned column quad
    // stays contiguous: stage float4 -> v4bf.
    for (int t = tid; t < ROWS * (DIM / 4); t += 256) {
        int row = t >> 6;                 // 64 quads per row
        int col = (t & 63) * 4;
        float4 v = *(const float4*)(x + (size_t)(row0 + row) * DIM + col);
        int kt = col >> 5, j = col & 31;
        int pj = j;
        v4bf p;
        p.x = (__bf16)v.x; p.y = (__bf16)v.y; p.z = (__bf16)v.z; p.w = (__bf16)v.w;
        *(v4bf*)(&xt[row * LDS_STRIDE + kt * 32 + pj]) = p;   // 8-B aligned
    }
    __syncthreads();

    int wave = tid >> 5, lane = tid & 31;
    int half = lane >> 4, l16 = lane & 15;

    // Preload the 16 (attn, graph) pairs this lane's D elements touch
    // (rows rt*16 + 8*half + r) -- reused across both n-tiles; kills the
    // per-element ds reads in the epilogue.
    float att[16];
    int   garr[16];
    #pragma unroll
    for (int i = 0; i < 16; ++i) {
        int row = (i >> 3) * 16 + 8 * half + (i & 7);
        att[i]  = s_attn[row];
        garr[i] = s_g[row];
    }

    // Common case (~74% of 32-row tiles with ~122-node avg graphs): the whole
    // tile belongs to one graph -> block-uniform scalar branch, one atomic per
    // lane per n-tile. Ragged fallback: one unconditional atomic per element
    // (branch-free; z is L2-resident).
    bool uniform = (s_g[0] == s_g[ROWS - 1]);
    int  g0      = s_g[0];

    #pragma unroll
    for (int nti = 0; nti < 2; ++nti) {
        int nt   = wave * 2 + nti;          // 8 waves x 2 = 16 N-tiles = 256 cols
        int ncol = nt * 16 + l16;
        float bias = value_b[ncol];

        // hoist all 8 K-step B operands for this n-tile (reused by both row-tiles)
        v16bf barr[8];
        #pragma unroll
        for (int kt = 0; kt < 8; ++kt)
            barr[kt] = *(const v16bf*)(pB + ((kt * 16 + nt) * 32 + lane) * 16);

        float accum = 0.0f;

        #pragma unroll
        for (int rt = 0; rt < 2; ++rt) {
            v8f acc;
            #pragma unroll
            for (int r = 0; r < 8; ++r) acc[r] = bias;

            int arow = rt * 16 + l16;       // A: lane holds matrix row M = lane&15
            const __bf16* abase = &xt[arow * LDS_STRIDE + half * 16];

            #pragma unroll
            for (int kt = 0; kt < 8; ++kt) {
                union { v16bf v; v8bf h[2]; } au;
                au.h[0] = *(const v8bf*)(abase + kt * 32);       // ds_load_b128
                au.h[1] = *(const v8bf*)(abase + kt * 32 + 8);   // ds_load_b128
                acc = __builtin_amdgcn_wmma_f32_16x16x32_bf16(
                        false, au.v, false, barr[kt], (short)0, acc, false, false);
            }

            // D layout: VGPR r holds row (r + 8*half), col = lane&15 of the tile
            if (uniform) {
                float s = 0.0f;
                #pragma unroll
                for (int r = 0; r < 8; ++r) {
                    float d = acc[r];
                    float v = d * fast_rcp(1.0f + __expf(-d));   // silu
                    s += v * att[rt * 8 + r];
                }
                accum += s;
            } else {
                #pragma unroll
                for (int r = 0; r < 8; ++r) {
                    int i = rt * 8 + r;
                    float d = acc[r];
                    float v = d * fast_rcp(1.0f + __expf(-d));   // silu
                    atomicAdd(&z[garr[i] * OUTD + ncol], v * att[i]);
                }
            }
        }
        if (uniform) {
            atomicAdd(&z[g0 * OUTD + ncol], accum);
        }
    }
}

// ---------------- launch ----------------
extern "C" void kernel_launch(void* const* d_in, const int* in_sizes, int n_in,
                              void* d_out, int out_size, void* d_ws, size_t ws_size,
                              hipStream_t stream) {
    // inputs: x, batch, n_graphs, gate_w, gate_b, value_w, value_b
    const float* x       = (const float*)d_in[0];
    const int*   batch   = (const int*)  d_in[1];
    const float* gate_w  = (const float*)d_in[3];
    const float* gate_b  = (const float*)d_in[4];
    const float* value_w = (const float*)d_in[5];
    const float* value_b = (const float*)d_in[6];

    // outputs: z [4096*256] then attn [500000]
    float* z    = (float*)d_out;
    float* attn = (float*)d_out + (size_t)NGRAPH * OUTD;

    // workspace layout
    float* gates = (float*)d_ws;                 // N
    float* expg  = gates + N_NODES;              // N
    float* gmax  = expg + N_NODES;               // NGRAPH
    float* gsum  = gmax + NGRAPH;                // NGRAPH
    __bf16* pB   = (__bf16*)(gsum + NGRAPH);     // 65536 bf16 (32-B aligned offset)

    init_kernel<<<(NGRAPH * OUTD + 255) / 256, 256, 0, stream>>>(z, gmax, gsum);
    pack_b_kernel<<<(DIM * OUTD) / 256, 256, 0, stream>>>(value_w, pB);
    gates_max_kernel<<<1024, 256, 0, stream>>>(x, batch, gate_w, gate_b, gates, gmax);
    exp_sum_kernel<<<(N_NODES + 255) / 256, 256, 0, stream>>>(batch, gates, gmax, expg, gsum);
    fused_pool_kernel<<<N_NODES / ROWS, 256, 0, stream>>>(
        x, batch, value_b, pB, expg, gsum, z, attn);
}
